// SDPA_78572131713603
// MI455X (gfx1250) — matrix operand order repaired and checked by reading, not compiled, not run on hardware
//
#include <hip/hip_runtime.h>

// ---------------------------------------------------------------------------
// SDPA (QKV proj + attention) for MI455X / gfx1250, wave32 + WMMA bf16 + TDM.
//
// Pipeline (intermediates L2-resident on 192MB L2):
//   0) convert x, Wq, Wk, Wv  f32 -> bf16 (raw u16) in workspace
//   1) q/k/v = x @ W^T + b     (bf16 WMMA GEMM, bf16 out)        3 launches
//   2) vT = transpose(v)       (reuses dead x-bf16 buffer)
//   3) scores = (q @ k^T)/32   (bf16 WMMA GEMM, f32 out)
//   4) row softmax, probs written bf16 in-place over scores rows
//   5) out = probs @ vT^T      (bf16 WMMA GEMM, f32 out -> d_out)
//
// GEMM: BM=256, BN=128, BK=32; 8 wave32 waves as 4(M)x2(N); each wave 64x64 =
// 4x4 WMMA 16x16x32 bf16 (16 wmma / K-step, 1.0 ds_load_b128 per wmma).
// Global->LDS staging via the Tensor Data Mover (tensor_load_to_lds,
// TENSORcnt), double-buffered issue-ahead-1; one s_barrier per K-step.
// ---------------------------------------------------------------------------

typedef __attribute__((ext_vector_type(16))) __bf16 v16bf;
typedef __attribute__((ext_vector_type(8)))  __bf16 v8bf;
typedef __attribute__((ext_vector_type(8)))  float  v8f;
typedef __attribute__((ext_vector_type(8)))  unsigned short u16x8;
typedef __attribute__((ext_vector_type(4)))  unsigned int   u32x4;
typedef __attribute__((ext_vector_type(8)))  int            i32x8;
typedef __attribute__((ext_vector_type(4)))  int            i32x4;
typedef unsigned short u16;

constexpr int BM  = 256;       // block tile M
constexpr int BN  = 128;       // block tile N
constexpr int BK  = 32;        // K step == WMMA K
constexpr int PAD = 8;         // LDS row pad (elems) -> 80B pitch, 16B aligned
constexpr int BKP = BK + PAD;
constexpr int TPB = 256;       // 8 wave32 waves
constexpr int NBUF = 2;        // double-buffered LDS tiles (60KB total)

#if defined(__has_builtin)
#if __has_builtin(__builtin_amdgcn_tensor_load_to_lds) && \
    __has_builtin(__builtin_amdgcn_s_wait_tensorcnt)
#define USE_TDM 1
#endif
#endif

__device__ __forceinline__ u16 f32_to_bf16(float f) {
  unsigned u = __float_as_uint(f);
  u += 0x7FFFu + ((u >> 16) & 1u);          // round to nearest even
  return (u16)(u >> 16);
}

__device__ __forceinline__ v16bf cat8(v8bf lo, v8bf hi) {
  return __builtin_shufflevector(lo, hi, 0,1,2,3,4,5,6,7,8,9,10,11,12,13,14,15);
}

__global__ void convert_f32_to_bf16(const float* __restrict__ in,
                                    u16* __restrict__ out, int n) {
  int i = blockIdx.x * blockDim.x + threadIdx.x;
  int stride = gridDim.x * blockDim.x;
  for (; i < n; i += stride) out[i] = f32_to_bf16(in[i]);
}

// 32x32 LDS tile transpose: out[c][r] = in[r][c], bf16. block (32,8).
__global__ __launch_bounds__(256)
void transpose_bf16(const u16* __restrict__ in, u16* __restrict__ out,
                    int R, int C) {
  __shared__ u16 t[32][33];
  const int r0 = blockIdx.x * 32, c0 = blockIdx.y * 32;
  const int tx = threadIdx.x, ty = threadIdx.y;
  in  += (size_t)blockIdx.z * R * C;
  out += (size_t)blockIdx.z * R * C;
  #pragma unroll
  for (int i = 0; i < 4; ++i)
    t[ty + i * 8][tx] = in[(size_t)(r0 + ty + i * 8) * C + c0 + tx];
  __syncthreads();
  #pragma unroll
  for (int i = 0; i < 4; ++i)
    out[(size_t)(c0 + ty + i * 8) * R + r0 + tx] = t[tx][ty + i * 8];
}

#ifdef USE_TDM
// Issue a TDM 2D tile load: rows x BK bf16 from row-major global (pitch
// strideElems) into LDS at ldsOff, with TDM padding reproducing the +PAD
// element LDS row pitch (16 data DWORDs, then 4 pad DWORDs).
__device__ __forceinline__ void tdm_load_tile(const u16* gptr, unsigned ldsOff,
                                              int rows, int strideElems) {
  unsigned long long ga = (unsigned long long)(uintptr_t)gptr;
  u32x4 g0;
  g0[0] = 1u;                                  // count=1 (valid user desc)
  g0[1] = ldsOff;                              // lds_addr (bytes)
  g0[2] = (unsigned)ga;                        // global_addr[31:0]
  g0[3] = (unsigned)((ga >> 32) & 0x1FFFFFFu)  // global_addr[56:32]
        | (2u << 30);                          // type=2 ("image")
  i32x8 g1 = (i32x8)0;
  // data_size=1 (2B) | pad_enable | pad_interval=3 (16 DW) | pad_amount=3 (4 DW)
  g1[0] = (int)((1u << 16) | (1u << 20) | (3u << 22) | (3u << 25));
  g1[1] = (int)(((unsigned)BK & 0xFFFFu) << 16);        // tensor_dim0[15:0]
  g1[2] = (int)((((unsigned)BK >> 16) & 0xFFFFu)        // tensor_dim0[31:16]
        | (((unsigned)rows & 0xFFFFu) << 16));          // tensor_dim1[15:0]
  g1[3] = (int)((((unsigned)rows >> 16) & 0xFFFFu)      // tensor_dim1[31:16]
        | ((unsigned)BK << 16));                        // tile_dim0
  g1[4] = (int)((unsigned)rows & 0xFFFFu);              // tile_dim1
  g1[5] = (int)(unsigned)strideElems;                   // tensor_dim0_stride lo
  g1[6] = 0;                                            // stride hi / dim1_stride lo
  g1[7] = 0;
  i32x4 z4 = (i32x4)0;
  i32x8 z8 = (i32x8)0;                         // groups 2/3 unused (2D tensor)
  __builtin_amdgcn_tensor_load_to_lds(g0, g1, z4, z4, z8, 0);
}
#endif

// ---------------------------------------------------------------------------
// Tiled bf16 WMMA GEMM:  C = scale * (A @ B^T) + bias
//   A: [M x K] row-major bf16 (u16 raw), B: [N x K] row-major bf16
//   C: f32 or bf16 (OUT_BF16)
// ---------------------------------------------------------------------------
template<bool OUT_BF16, bool HAS_BIAS>
__global__ __launch_bounds__(TPB)
void gemm_bf16_wmma(const u16* __restrict__ A, const u16* __restrict__ B,
                    const float* __restrict__ bias, void* __restrict__ Cout,
                    int M, int N, int K, int lda, int ldb, int ldc,
                    long long sA, long long sB, long long sC, float scale)
{
  __shared__ u16 AsBuf[NBUF][BM][BKP];
  __shared__ u16 BsBuf[NBUF][BN][BKP];

  const int tid  = threadIdx.x;
  const int lane = tid & 31;
  const int wave = tid >> 5;
  const int l16  = lane & 15;
  const int half = lane >> 4;        // wave32 half (ISA 7.12.2 layouts)
  const int wm   = wave & 3;         // 4 waves along M, 64 rows each
  const int wn   = wave >> 2;        // 2 waves along N, 64 cols each

  const int rowStart = blockIdx.y * BM;
  const int colStart = blockIdx.x * BN;
  A += sA * blockIdx.z + (size_t)rowStart * lda;
  B += sB * blockIdx.z + (size_t)colStart * ldb;

  const int KT = K / BK;             // >= 2 for all our shapes
  v8f acc[4][4] = {};

#ifdef USE_TDM
  // --- preload tile 0 via the Tensor Data Mover (wave 0 issues) ---
  if (wave == 0) {
    tdm_load_tile(A, (unsigned)(uintptr_t)&AsBuf[0][0][0], BM, lda);
    tdm_load_tile(B, (unsigned)(uintptr_t)&BsBuf[0][0][0], BN, ldb);
  }
#endif

  for (int j = 0; j < KT; ++j) {
    const int bsel = j & 1;
#ifdef USE_TDM
    if (wave == 0)
      __builtin_amdgcn_s_wait_tensorcnt((short)0);  // tile j's A+B landed
    __syncthreads();                                // + everyone left tile j-1
    if (wave == 0 && j + 1 < KT) {                  // stream tile j+1 into the
      const int nsel = bsel ^ 1;                    // buffer tile j-1 vacated
      tdm_load_tile(A + (j + 1) * BK,
                    (unsigned)(uintptr_t)&AsBuf[nsel][0][0], BM, lda);
      tdm_load_tile(B + (j + 1) * BK,
                    (unsigned)(uintptr_t)&BsBuf[nsel][0][0], BN, ldb);
    }
#else
    // fallback: synchronous cooperative staging (16B chunks)
    __syncthreads();
    const int k0 = j * BK;
    #pragma unroll
    for (int s = 0; s < 4; ++s) {                   // A: 256x32 = 1024 chunks
      int c = tid + s * TPB;
      int r = c >> 2, c8 = (c & 3) * 8;
      *(u16x8*)&AsBuf[bsel][r][c8] =
          *(const u16x8*)(A + (size_t)r * lda + k0 + c8);
    }
    #pragma unroll
    for (int s = 0; s < 2; ++s) {                   // B: 128x32 = 512 chunks
      int c = tid + s * TPB;
      int r = c >> 2, c8 = (c & 3) * 8;
      *(u16x8*)&BsBuf[bsel][r][c8] =
          *(const u16x8*)(B + (size_t)r * ldb + k0 + c8);
    }
    if (k0 + BK < K)
      __builtin_prefetch(A + (size_t)(tid & (BM - 1)) * lda + k0 + BK, 0, 1);
    __syncthreads();
#endif

    // ---- fragment loads per ISA 7.12.2 16-bit layouts ----
    const u16 (*As)[BKP] = AsBuf[bsel];
    const u16 (*Bs)[BKP] = BsBuf[bsel];
    v16bf af[4], bfr[4];
    #pragma unroll
    for (int mi = 0; mi < 4; ++mi) {   // A: row l16, K chunks half*8 / 16+half*8
      const u16* p = &As[wm * 64 + mi * 16 + l16][0];
      af[mi] = cat8(*(const v8bf*)(p + half * 8),
                    *(const v8bf*)(p + 16 + half * 8));
    }
    #pragma unroll
    for (int ni = 0; ni < 4; ++ni) {   // B: col l16, K chunks half*8 / 16+half*8
      const u16* p = &Bs[wn * 64 + ni * 16 + l16][0];
      bfr[ni] = cat8(*(const v8bf*)(p + half * 8),
                     *(const v8bf*)(p + 16 + half * 8));
    }
    #pragma unroll
    for (int mi = 0; mi < 4; ++mi)
      #pragma unroll
      for (int ni = 0; ni < 4; ++ni)
        acc[mi][ni] = __builtin_amdgcn_wmma_f32_16x16x32_bf16(
            false, af[mi], false, bfr[ni], (short)0, acc[mi][ni],
            false, false);
  }

  // ---- epilogue: C layout VGPR v -> M = v + 8*half, N = lane&15 ----
  float* Cf = (float*)Cout;
  u16*   Cb = (u16*)Cout;
  #pragma unroll
  for (int mi = 0; mi < 4; ++mi) {
    #pragma unroll
    for (int ni = 0; ni < 4; ++ni) {
      int col = colStart + wn * 64 + ni * 16 + l16;
      float bv = HAS_BIAS ? bias[col] : 0.0f;
      #pragma unroll
      for (int v = 0; v < 8; ++v) {
        int row = rowStart + wm * 64 + mi * 16 + v + 8 * half;
        float val = acc[mi][ni][v] * scale + bv;
        size_t idx = (size_t)(sC * blockIdx.z) + (size_t)row * ldc + col;
        if (OUT_BF16) Cb[idx] = f32_to_bf16(val);
        else          Cf[idx] = val;
      }
    }
  }
}

// ---------------------------------------------------------------------------
// Row softmax over L=2048 f32, rewritten in place as bf16 (low half of each
// f32 row -> no cross-block aliasing; downstream GEMM uses lda = 2*L elems).
// ---------------------------------------------------------------------------
__global__ __launch_bounds__(256)
void softmax_row_inplace_bf16(float* __restrict__ scores, int L) {
  const int tid = threadIdx.x;
  float* rp = scores + (size_t)blockIdx.x * L;
  __shared__ float red[256];

  float v[8];
  float m = -3.0e38f;
  #pragma unroll
  for (int i = 0; i < 8; ++i) { v[i] = rp[tid + i * 256]; m = fmaxf(m, v[i]); }
  red[tid] = m; __syncthreads();
  for (int s = 128; s > 0; s >>= 1) {
    if (tid < s) red[tid] = fmaxf(red[tid], red[tid + s]);
    __syncthreads();
  }
  m = red[0]; __syncthreads();

  float sum = 0.f;
  #pragma unroll
  for (int i = 0; i < 8; ++i) { v[i] = __expf(v[i] - m); sum += v[i]; }
  red[tid] = sum; __syncthreads();
  for (int s = 128; s > 0; s >>= 1) {
    if (tid < s) red[tid] += red[tid + s];
    __syncthreads();
  }
  float inv = 1.0f / red[0];

  u16* op = (u16*)rp;
  #pragma unroll
  for (int i = 0; i < 8; ++i) op[tid + i * 256] = f32_to_bf16(v[i] * inv);
}

// ---------------------------------------------------------------------------
extern "C" void kernel_launch(void* const* d_in, const int* in_sizes, int n_in,
                              void* d_out, int out_size, void* d_ws, size_t ws_size,
                              hipStream_t stream)
{
  (void)in_sizes; (void)n_in; (void)out_size; (void)ws_size;
  constexpr int Bz = 4, S = 2048, D = 1024;

  const float* x  = (const float*)d_in[0];
  const float* Wq = (const float*)d_in[1];
  const float* bq = (const float*)d_in[2];
  const float* Wk = (const float*)d_in[3];
  const float* bk = (const float*)d_in[4];
  const float* Wv = (const float*)d_in[5];
  const float* bv = (const float*)d_in[6];
  float* out = (float*)d_out;

  const size_t nX = (size_t)Bz * S * D;   // 8,388,608
  const size_t nW = (size_t)D * D;        // 1,048,576

  // workspace layout (bf16 raw u16 unless noted); ~134 MB.
  // xbf is dead after the projections and is reused for vT.
  u16* xbf = (u16*)d_ws;
  u16* vT  = xbf;                          // [Bz][D][S] after transpose
  u16* qb  = xbf + nX;
  u16* kb  = qb  + nX;
  u16* vb  = kb  + nX;
  u16* wqb = vb  + nX;
  u16* wkb = wqb + nW;
  u16* wvb = wkb + nW;
  float* scores = (float*)(wvb + nW);      // [Bz][S][S] f32, softmax in place

  // 0) precision drop to bf16 (~50 MB traffic, ~2us at 23.3 TB/s)
  convert_f32_to_bf16<<<4096, 256, 0, stream>>>(x,  xbf, (int)nX);
  convert_f32_to_bf16<<<1024, 256, 0, stream>>>(Wq, wqb, (int)nW);
  convert_f32_to_bf16<<<1024, 256, 0, stream>>>(Wk, wkb, (int)nW);
  convert_f32_to_bf16<<<1024, 256, 0, stream>>>(Wv, wvb, (int)nW);

  // 1) q/k/v = x @ W^T + b     (M=8192, N=1024, K=1024)
  dim3 gProj(D / BN, (Bz * S) / BM, 1);
  gemm_bf16_wmma<true, true><<<gProj, TPB, 0, stream>>>(
      xbf, wqb, bq, qb, Bz * S, D, D, D, D, D, 0, 0, 0, 1.0f);
  gemm_bf16_wmma<true, true><<<gProj, TPB, 0, stream>>>(
      xbf, wkb, bk, kb, Bz * S, D, D, D, D, D, 0, 0, 0, 1.0f);
  gemm_bf16_wmma<true, true><<<gProj, TPB, 0, stream>>>(
      xbf, wvb, bv, vb, Bz * S, D, D, D, D, D, 0, 0, 0, 1.0f);

  // 2) vT[b][d][s] = v[b][s][d]  (x buffer is dead now)
  transpose_bf16<<<dim3(S / 32, D / 32, Bz), dim3(32, 8), 0, stream>>>(
      vb, vT, S, D);

  // 3) scores = (q @ k^T) * 1/sqrt(D)   (per batch: M=N=2048, K=1024)
  dim3 gScore(S / BN, S / BM, Bz);
  gemm_bf16_wmma<false, false><<<gScore, TPB, 0, stream>>>(
      qb, kb, nullptr, scores, S, S, D, D, D, S,
      (long long)S * D, (long long)S * D, (long long)S * S, 0.03125f);

  // 4) softmax rows -> bf16 probs in place
  softmax_row_inplace_bf16<<<Bz * S, 256, 0, stream>>>(scores, S);

  // 5) out = probs @ vT^T   (per batch: M=2048, N=1024, K=2048)
  //    probs lda = 2*S u16 elems (bf16 packed in low half of f32 rows)
  dim3 gAV(D / BN, S / BM, Bz);
  gemm_bf16_wmma<false, false><<<gAV, TPB, 0, stream>>>(
      (const u16*)scores, vT, nullptr, out, S, D, S, 2 * S, S, D,
      (long long)2 * S * S, (long long)D * S, (long long)S * D, 1.0f);
}
